// SelfAttention_88381837017520
// MI455X (gfx1250) — matrix operand order repaired and checked by reading, not compiled, not run on hardware
//
#include <hip/hip_runtime.h>

// ---------------------------------------------------------------------------
// Causal single-head self-attention, CDNA5 (gfx1250), wave32.
// B=4, S=2048, E=A=1024, fp32 in/out. All GEMMs on V_WMMA_F32_16X16X32_BF16.
// Unified GEMM: 2x64-wide K chunks per trip, ping-pong LDS buffers filled by
// GLOBAL_LOAD_ASYNC_TO_LDS_B128 (ASYNCcnt), ping-pong A fragments in
// registers, sched_group_barrier-pipelined DS reads across the WMMA stream.
// Causality via block-level tile skipping (~6% overcompute vs triangular).
// ---------------------------------------------------------------------------

typedef __attribute__((ext_vector_type(16))) __bf16          v16bf;
typedef __attribute__((ext_vector_type(8)))  float           v8f;
typedef __attribute__((ext_vector_type(8)))  unsigned short  u16x8;

constexpr int Bn = 4;      // batch
constexpr int Sn = 2048;   // sequence
constexpr int En = 1024;   // embed
constexpr int An = 1024;   // attn dim

// LDS B-tile: 64 rows x 64 bf16, rows padded to 72 elems (144B, 16B-aligned).
// Column-fragment reads: 16 consecutive rows -> (36*l mod 64) banks, distinct.
// Two buffers = 18 KB.
constexpr int LDS_ROW = 72;

#if defined(__has_builtin) && __has_builtin(__builtin_amdgcn_sched_group_barrier)
#define SCHED_GROUP(mask, size) __builtin_amdgcn_sched_group_barrier((mask), (size), 0)
#else
#define SCHED_GROUP(mask, size)
#endif
// scheduling class masks: 0x008 = MFMA/WMMA, 0x100 = DS read

// float -> bf16, round-to-nearest-even
__device__ __forceinline__ unsigned short f2bf(float f) {
    unsigned int u = __float_as_uint(f);
    u += 0x7FFFu + ((u >> 16) & 1u);
    return (unsigned short)(u >> 16);
}
__device__ __forceinline__ float bf2f(unsigned short h) {
    return __uint_as_float(((unsigned int)h) << 16);
}

// ---------------------------------------------------------------------------
// WMMA fragment helpers per CDNA5 ISA VGPR layouts (CDNA5 ISA):
//  A (16x32 MxK): lane m holds row M=m (lanes 16-31: +8 K offset);
//    e=0..7 -> K=koff+e, e=8..15 -> K=koff+16+(e-8)   => two b128 loads.
//  B (32x16 KxN): lane n holds column N=n&15, K split as for A (koff=8*(lane>=16)); our B
//    matrices are row-major [N,K] so each lane does two b128 reads (K+0..7, K+16..23).
//  C/D (16x16 f32): VGPR r, lane l -> M = r + 8*(l>=16), N = l&15.
// ---------------------------------------------------------------------------

// A fragment at per-lane base pointer p (already includes row and lane-half).
__device__ __forceinline__ v16bf load_a_frag(const unsigned short* p) {
    union { v16bf v; u16x8 h[2]; } u;
    u.h[0] = *(const u16x8*)(p);
    u.h[1] = *(const u16x8*)(p + 16);
    return u.v;
}

// B fragment from the padded LDS tile: tile j (16 N-cols), K-half h (0/1).
__device__ __forceinline__ v16bf lds_b_frag(const unsigned short* sbuf, int j, int h, int lane) {
    const unsigned short* p =
        sbuf + (j * 16 + (lane & 15)) * LDS_ROW + h * 32 + ((lane >> 4) << 3);
    union { v16bf v; u16x8 h2[2]; } u;
    u.h2[0] = *(const u16x8*)(p);
    u.h2[1] = *(const u16x8*)(p + 16);
    return u.v;
}

// Async 16B global->LDS copy with compile-time byte offset on the address.
template<int OFFB>
__device__ __forceinline__ void async_cp16(unsigned int lds, const unsigned short* g) {
    asm volatile("global_load_async_to_lds_b128 %0, %1, off offset:%2"
                 :: "v"(lds), "v"(g), "i"(OFFB) : "memory");
}
// Wait for our async copies. No memory clobber: ordering vs the copy asm is
// via volatile-asm program order, and the subsequent __syncthreads() fences
// the DS reads — this keeps outstanding global loads (A prefetch) unpunished.
__device__ __forceinline__ void wait_async0() {
    asm volatile("s_wait_asynccnt 0x0");
}

// 8 WMMAs of one 64-wide K chunk; sched-group pipeline: 8 DS reads ahead of
// the first WMMA pair, remaining fragment loads interleaved with WMMAs.
__device__ __forceinline__ void chunk_mma(v8f c[4], const unsigned short* sb,
                                          v16bf aLo, v16bf aHi, int lane) {
    v16bf b0 = lds_b_frag(sb, 0, 0, lane);
    v16bf b1 = lds_b_frag(sb, 1, 0, lane);
    v16bf b2 = lds_b_frag(sb, 2, 0, lane);
    v16bf b3 = lds_b_frag(sb, 3, 0, lane);
    c[0] = __builtin_amdgcn_wmma_f32_16x16x32_bf16(false, aLo, false, b0, (short)0, c[0], false, false);
    c[1] = __builtin_amdgcn_wmma_f32_16x16x32_bf16(false, aLo, false, b1, (short)0, c[1], false, false);
    v16bf b4 = lds_b_frag(sb, 0, 1, lane);
    v16bf b5 = lds_b_frag(sb, 1, 1, lane);
    c[2] = __builtin_amdgcn_wmma_f32_16x16x32_bf16(false, aLo, false, b2, (short)0, c[2], false, false);
    c[3] = __builtin_amdgcn_wmma_f32_16x16x32_bf16(false, aLo, false, b3, (short)0, c[3], false, false);
    v16bf b6 = lds_b_frag(sb, 2, 1, lane);
    v16bf b7 = lds_b_frag(sb, 3, 1, lane);
    c[0] = __builtin_amdgcn_wmma_f32_16x16x32_bf16(false, aHi, false, b4, (short)0, c[0], false, false);
    c[1] = __builtin_amdgcn_wmma_f32_16x16x32_bf16(false, aHi, false, b5, (short)0, c[1], false, false);
    c[2] = __builtin_amdgcn_wmma_f32_16x16x32_bf16(false, aHi, false, b6, (short)0, c[2], false, false);
    c[3] = __builtin_amdgcn_wmma_f32_16x16x32_bf16(false, aHi, false, b7, (short)0, c[3], false, false);
    // Pin the pipeline shape for this chunk (16 DS reads, 8 WMMAs):
    SCHED_GROUP(0x100, 8);  // b0..b3 (8 ds_load_b128)
    SCHED_GROUP(0x008, 2);  // wmma c0,c1 (aLo)
    SCHED_GROUP(0x100, 4);  // b4,b5
    SCHED_GROUP(0x008, 2);  // wmma c2,c3 (aLo)
    SCHED_GROUP(0x100, 4);  // b6,b7
    SCHED_GROUP(0x008, 4);  // wmma c0..c3 (aHi)
}

// ---------------------------------------------------------------------------
// fp32 -> bf16 bulk convert (vectorized x4)
// ---------------------------------------------------------------------------
__global__ __launch_bounds__(256)
void cvt_f32_bf16(const float* __restrict__ in, unsigned short* __restrict__ out, int n4) {
    int i = blockIdx.x * blockDim.x + threadIdx.x;
    if (i >= n4) return;
    float4 f = ((const float4*)in)[i];
    ushort4 o;
    o.x = f2bf(f.x); o.y = f2bf(f.y); o.z = f2bf(f.z); o.w = f2bf(f.w);
    ((ushort4*)out)[i] = o;
}

// ---------------------------------------------------------------------------
// Unified bf16 WMMA GEMM:  D[M,N] = scale * A[M,K] * B[N,K]^T
// Block = 8 waves = 128(M) x 64(N); wave = 16 x 64; K consumed 128 per trip
// (two 64 chunks, ping-pong buffers/registers). K must be a multiple of 128.
// MODE 0: bf16 row-major   MODE 1: bf16 transposed Vt[b][n][s]   MODE 2: f32.
// CAUSAL: skip whole blocks above the diagonal (uniform exit, pre-barrier).
// ---------------------------------------------------------------------------
template<int MODE, bool CAUSAL>
__global__ __launch_bounds__(256)
void gemm_bf16(const unsigned short* __restrict__ Am, int lda, long long sA,
               const unsigned short* __restrict__ Bm, int ldb, long long sB,
               void* __restrict__ Out, int ldo, long long sO,
               int M, int N, int K, float scale) {
    __shared__ __align__(16) unsigned short sB2[2][64 * LDS_ROW];

    if (CAUSAL) {
        // keys start at 64*bx, max query row is 128*by+127
        if (blockIdx.x * 64 > blockIdx.y * 128 + 127) return;
    }

    const int tid  = threadIdx.x;
    const int lane = tid & 31;
    const int wave = __builtin_amdgcn_readfirstlane(tid >> 5);  // scalar wave id
    const int z    = blockIdx.z;
    const unsigned short* A = Am + (size_t)z * sA;
    const unsigned short* B = Bm + (size_t)z * sB;
    const int m0 = blockIdx.y * 128 + wave * 16;
    const int n0 = blockIdx.x * 64;

    // Per-thread strided pointers (advance by constants; offsets in immediates)
    const unsigned short* pa =
        A + (size_t)(m0 + (lane & 15)) * lda + ((lane >> 4) << 3);
    const unsigned short* pb =
        B + (size_t)(n0 + (tid >> 2)) * ldb + (tid & 3) * 8;
    // LDS destinations: byte offset == low 32 bits of generic pointer
    // (ISA 10.2: LDS aperture addresses truncate to addr[31:0]).
    const unsigned int lds0 =
        (unsigned int)(size_t)(&sB2[0][0]) + (tid >> 2) * (LDS_ROW * 2) + (tid & 3) * 16;
    const unsigned int lds1 = lds0 + 64 * LDS_ROW * 2;

    // prologue: stage chunk 0 into buffer 0, load first two A fragments
    async_cp16<0>(lds0, pb);
    async_cp16<64>(lds0 + 64, pb);
    v8f c[4] = {};
    v16bf a0 = load_a_frag(pa);
    v16bf a1 = load_a_frag(pa + 32);

    for (int k0 = 0; k0 < K; k0 += 128) {
        // ---- chunk 0: compute from buf0 / a0,a1; prefetch k0+64 ----
        wait_async0();
        __syncthreads();
        async_cp16<128>(lds1, pb);                 // k0+64 < K always
        async_cp16<192>(lds1 + 64, pb);
        v16bf a2 = load_a_frag(pa + 64);
        v16bf a3 = load_a_frag(pa + 96);
        chunk_mma(c, sB2[0], a0, a1, lane);

        // ---- chunk 1: compute from buf1 / a2,a3; prefetch k0+128 ----
        wait_async0();
        __syncthreads();
        const bool more = (k0 + 128 < K);
        if (more) {
            async_cp16<256>(lds0, pb);
            async_cp16<320>(lds0 + 64, pb);
            a0 = load_a_frag(pa + 128);
            a1 = load_a_frag(pa + 160);
        }
        chunk_mma(c, sB2[1], a2, a3, lane);
        pa += 128;
        pb += 128;      // 256B: matches the offset:256/320 of the next trip
    }

    // Store. (For CAUSAL, k>q entries are garbage-but-finite scores; the
    // softmax kernel masks them out and overwrites them with 0.)
    const int cn    = lane & 15;
    const int rbase = (lane >> 4) << 3;
#pragma unroll
    for (int j = 0; j < 4; ++j) {
#pragma unroll
        for (int r = 0; r < 8; ++r) {
            const int row = m0 + rbase + r;
            const int col = n0 + j * 16 + cn;
            const float v = c[j][r] * scale;
            if (MODE == 0) {
                ((unsigned short*)Out)[(size_t)z * sO + (size_t)row * ldo + col] = f2bf(v);
            } else if (MODE == 1) {
                const int bb = row >> 11;          // row / Sn
                const int s  = row & (Sn - 1);     // row % Sn
                ((unsigned short*)Out)[((size_t)bb * An + col) * Sn + s] = f2bf(v);
            } else {
                ((float*)Out)[(size_t)z * sO + (size_t)row * ldo + col] = v;
            }
        }
    }
}

// ---------------------------------------------------------------------------
// Causal row softmax in place on bf16 P (fp32 math). One wave per row;
// 8-wide vector passes with per-element causal masking. Values beyond the
// diagonal are finite bf16 (unwritten 0xAAAA poison or stale data), so the
// masked selects are NaN-safe. Writes all S entries (0 beyond diagonal).
// ---------------------------------------------------------------------------
__global__ __launch_bounds__(256)
void softmax_causal(unsigned short* __restrict__ P) {
    const int lane = threadIdx.x & 31;
    const int wave = __builtin_amdgcn_readfirstlane((int)(threadIdx.x >> 5));
    const int row  = blockIdx.x * 8 + wave;   // 0 .. B*S-1
    const int q    = row & (Sn - 1);
    unsigned short* Pr = P + (size_t)(row >> 11) * Sn * Sn + (size_t)q * Sn;
    const int nv = q + 1;

    float m = -3.0e38f;
    for (int base = lane * 8; base < Sn; base += 256) {
        u16x8 vch = *(const u16x8*)(Pr + base);
#pragma unroll
        for (int e = 0; e < 8; ++e) {
            const float v = bf2f(vch[e]);
            m = fmaxf(m, (base + e < nv) ? v : -3.0e38f);
        }
    }
#pragma unroll
    for (int off = 16; off > 0; off >>= 1) m = fmaxf(m, __shfl_xor(m, off, 32));

    float s = 0.0f;
    for (int base = lane * 8; base < Sn; base += 256) {
        u16x8 vch = *(const u16x8*)(Pr + base);
#pragma unroll
        for (int e = 0; e < 8; ++e) {
            const float v = bf2f(vch[e]);
            s += (base + e < nv) ? __expf(v - m) : 0.0f;
        }
    }
#pragma unroll
    for (int off = 16; off > 0; off >>= 1) s += __shfl_xor(s, off, 32);

    const float inv = 1.0f / s;
    for (int base = lane * 8; base < Sn; base += 256) {
        u16x8 vch = *(const u16x8*)(Pr + base);
        u16x8 o;
#pragma unroll
        for (int e = 0; e < 8; ++e) {
            const float v = bf2f(vch[e]);
            o[e] = f2bf((base + e < nv) ? __expf(v - m) * inv : 0.0f);
        }
        *(u16x8*)(Pr + base) = o;
    }
}

// ---------------------------------------------------------------------------
// Host-side orchestration (graph-capture safe: kernel launches only).
// Workspace (bf16): xb | Wq/Wk/Wv/Wo | Q | K | Vt[b][a][s] | P[b][s][s] | AO
// ~120 MB total; everything stays L2-resident (192 MB).
// ---------------------------------------------------------------------------
extern "C" void kernel_launch(void* const* d_in, const int* in_sizes, int n_in,
                              void* d_out, int out_size, void* d_ws, size_t ws_size,
                              hipStream_t stream) {
    const float* x  = (const float*)d_in[0];
    const float* Wq = (const float*)d_in[1];
    const float* Wk = (const float*)d_in[2];
    const float* Wv = (const float*)d_in[3];
    const float* Wo = (const float*)d_in[4];
    // d_in[5] = padding_mask: all-valid; causal mask applied structurally.

    char*  ws  = (char*)d_ws;
    size_t off = 0;
    auto take = [&](size_t elems) {
        unsigned short* p = (unsigned short*)(ws + off);
        off += ((elems * 2 + 255) & ~(size_t)255);
        return p;
    };
    unsigned short* xb  = take((size_t)Bn * Sn * En);
    unsigned short* wqb = take((size_t)An * En);
    unsigned short* wkb = take((size_t)An * En);
    unsigned short* wvb = take((size_t)An * En);
    unsigned short* wob = take((size_t)An * An);
    unsigned short* Qb  = take((size_t)Bn * Sn * An);
    unsigned short* Kb  = take((size_t)Bn * Sn * An);
    unsigned short* Vt  = take((size_t)Bn * An * Sn);
    unsigned short* P   = take((size_t)Bn * Sn * Sn);
    unsigned short* AO  = take((size_t)Bn * Sn * An);

    const dim3 blk(256);

    // 1) bf16 conversions
    {
        int n4 = (Bn * Sn * En) / 4;
        cvt_f32_bf16<<<(n4 + 255) / 256, blk, 0, stream>>>(x, xb, n4);
        int w4 = (An * En) / 4;
        cvt_f32_bf16<<<(w4 + 255) / 256, blk, 0, stream>>>(Wq, wqb, w4);
        cvt_f32_bf16<<<(w4 + 255) / 256, blk, 0, stream>>>(Wk, wkb, w4);
        cvt_f32_bf16<<<(w4 + 255) / 256, blk, 0, stream>>>(Wv, wvb, w4);
        cvt_f32_bf16<<<(w4 + 255) / 256, blk, 0, stream>>>(Wo, wob, w4);
    }

    // 2) Q/K/V projections: [B*S,E] x [A,E]^T -> [B*S,A]
    const dim3 gproj(An / 64, (Bn * Sn) / 128, 1);
    gemm_bf16<0, false><<<gproj, blk, 0, stream>>>(xb, En, 0, wqb, En, 0,
                                                   Qb, An, 0, Bn * Sn, An, En, 1.0f);
    gemm_bf16<0, false><<<gproj, blk, 0, stream>>>(xb, En, 0, wkb, En, 0,
                                                   Kb, An, 0, Bn * Sn, An, En, 1.0f);
    gemm_bf16<1, false><<<gproj, blk, 0, stream>>>(xb, En, 0, wvb, En, 0,
                                                   Vt, An, 0, Bn * Sn, An, En, 1.0f);

    // 3) causal scores (per batch): P = (Q K^T) / sqrt(A), bf16
    const dim3 gsc(Sn / 64, Sn / 128, Bn);
    gemm_bf16<0, true><<<gsc, blk, 0, stream>>>(Qb, An, (long long)Sn * An,
                                                Kb, An, (long long)Sn * An,
                                                P, Sn, (long long)Sn * Sn,
                                                Sn, Sn, An, 0.03125f);

    // 4) causal softmax in place
    softmax_causal<<<(Bn * Sn) / 8, blk, 0, stream>>>(P);

    // 5) O = P * V (per batch): [S,S] x Vt[A,S]^T -> [S,A]
    const dim3 gav(An / 64, Sn / 128, Bn);
    gemm_bf16<0, false><<<gav, blk, 0, stream>>>(P, Sn, (long long)Sn * Sn,
                                                 Vt, Sn, (long long)An * Sn,
                                                 AO, An, (long long)Sn * An,
                                                 Sn, An, Sn, 1.0f);

    // 6) final projection, fp32 to d_out: [B*S,A] x [A,A]^T -> [B*S,A]
    gemm_bf16<2, false><<<gproj, blk, 0, stream>>>(AO, An, 0, wob, An, 0,
                                                   d_out, An, 0, Bn * Sn, An, An, 1.0f);
}
